// TripletLossWithMining_60825326846556
// MI455X (gfx1250) — matrix operand order repaired and checked by reading, not compiled, not run on hardware
//
#include <hip/hip_runtime.h>
#include <math.h>

typedef _Float16 half_t;
typedef __attribute__((ext_vector_type(16))) _Float16 v16h;
typedef __attribute__((ext_vector_type(8)))  _Float16 v8h;
typedef __attribute__((ext_vector_type(8)))  float    v8f;

#define N_ROWS 2048
#define DIM    1024
#define MARGIN 0.3f

// ---------------------------------------------------------------------------
// Kernel 1: L2-normalize each row (fp32) and emit an f16 copy for WMMA.
// Block 0 thread 0 also zeroes the global accumulators (runs before strip
// kernel on the same stream, so ordering is guaranteed even under graphs).
// ---------------------------------------------------------------------------
__global__ __launch_bounds__(256) void tl_normalize(const float* __restrict__ emb,
                                                    half_t* __restrict__ E,
                                                    float* gsum, unsigned* gcnt) {
    __shared__ float red[256];
    const int row = blockIdx.x;
    const float* src = emb + (size_t)row * DIM;
    float x[4];
    float s = 0.f;
#pragma unroll
    for (int i = 0; i < 4; ++i) {
        x[i] = src[threadIdx.x + 256 * i];
        s += x[i] * x[i];
    }
    red[threadIdx.x] = s;
    __syncthreads();
    for (int off = 128; off > 0; off >>= 1) {
        if (threadIdx.x < off) red[threadIdx.x] += red[threadIdx.x + off];
        __syncthreads();
    }
    const float inv = 1.0f / fmaxf(sqrtf(red[0]), 1e-12f);
    half_t* dst = E + (size_t)row * DIM;
#pragma unroll
    for (int i = 0; i < 4; ++i)
        dst[threadIdx.x + 256 * i] = (half_t)(x[i] * inv);
    if (row == 0 && threadIdx.x == 0) { *gsum = 0.f; *gcnt = 0u; }
}

// ---------------------------------------------------------------------------
// Kernel 2: one workgroup per 16-anchor strip, 8 wave32s.
//  Stage 1: anchors staged in LDS (32KB); the fully-unrolled K loop lets the
//           compiler promote all 32 A-fragments into the extended VGPR file
//           (round-1 codegen), streaming B tiles with v_wmma_f32_16x16x32_f16.
//  Stage 2: semi-hard mining over the LDS distance strip; labels read from
//           global (L2-resident) to keep LDS at exactly 160KB.
// LDS: 32KB anchors(f16) + 128KB pd(f32) = 160KB -> 2 workgroups per 320KB WGP.
// ---------------------------------------------------------------------------
__global__ __launch_bounds__(256) void tl_strip(const half_t* __restrict__ E,
                                                const int* __restrict__ labels,
                                                float* gsum, unsigned* gcnt) {
    __shared__ half_t sA[16 * DIM];
    __shared__ float  sPD[16 * N_ROWS];

    const int tid  = threadIdx.x;
    const int lane = tid & 31;
    const int wid  = tid >> 5;
    const int row0 = blockIdx.x * 16;

    // Stage 0: cooperative copy of the anchor strip into LDS.
    {
        const v8h* src = (const v8h*)(E + (size_t)row0 * DIM);
        v8h* dst = (v8h*)sA;
        for (int i = tid; i < 16 * DIM / 8; i += 256) dst[i] = src[i];
    }
    __syncthreads();

    // Stage 1: Gram tiles -> distances.
    // A-frag (16-bit A 16x32): lane m%16 = row; lanes<16 hold K 0..15,
    // lanes>=16 hold K 16..31.
    // B-frag (16-bit B 32x16): lane n%16 = col; lanes<16 K 0..15, lanes>=16 K 16..31.
    const half_t* abase = sA + (size_t)(lane & 15) * DIM + ((lane >> 4) << 4);
    for (int jt = wid; jt < N_ROWS / 16; jt += 8) {
        v8f c = {};
        const half_t* bbase = E + (size_t)(jt * 16 + (lane & 15)) * DIM + ((lane >> 4) << 4);
#pragma unroll
        for (int kc = 0; kc < DIM / 32; ++kc) {
            v8h lo = *(const v8h*)(abase + kc * 32);
            v8h hi = *(const v8h*)(abase + kc * 32 + 8);
            v16h a;
#pragma unroll
            for (int h = 0; h < 8; ++h) { a[h] = lo[h]; a[h + 8] = hi[h]; }
            v16h b = *(const v16h*)(bbase + kc * 32);
            c = __builtin_amdgcn_wmma_f32_16x16x32_f16(
                    false, a, false, b, (short)0, c, false, false);
        }
        // C layout: lanes 0-15 -> N=lane, M=vgpr; lanes 16-31 -> N=lane-16, M=vgpr+8
        const int n = lane & 15;
        const int mbase = (lane >> 4) * 8;
#pragma unroll
        for (int r = 0; r < 8; ++r) {
            float g  = c[r];
            float sq = fmaxf(2.f - 2.f * g, 1e-12f);   // clip(...,0) then safe sqrt
            sPD[(mbase + r) * N_ROWS + jt * 16 + n] = sqrtf(sq);
        }
    }
    __syncthreads();

    // Stage 2: mining. Each wave handles 2 anchor rows; labels from global/L2.
    for (int r = wid; r < 16; r += 8) {
        const int ga = row0 + r;
        const int alab = labels[ga];
        float hp = -__builtin_inff();
        bool hasPos = false, hasNeg = false;
        for (int col = lane; col < N_ROWS; col += 32) {
            const bool same = (labels[col] == alab);
            if (same) {
                if (col != ga) { hasPos = true; hp = fmaxf(hp, sPD[r * N_ROWS + col]); }
            } else {
                hasNeg = true;
            }
        }
#pragma unroll
        for (int off = 16; off; off >>= 1) hp = fmaxf(hp, __shfl_xor(hp, off, 32));
        hasPos = (__ballot(hasPos) != 0ull);
        hasNeg = (__ballot(hasNeg) != 0ull);

        float shn = __builtin_inff();
        bool hasSemi = false;
        for (int col = lane; col < N_ROWS; col += 32) {
            if (labels[col] != alab) {
                const float p = sPD[r * N_ROWS + col];
                if (p > hp) { hasSemi = true; shn = fminf(shn, p); }
            }
        }
#pragma unroll
        for (int off = 16; off; off >>= 1) shn = fminf(shn, __shfl_xor(shn, off, 32));
        hasSemi = (__ballot(hasSemi) != 0ull);

        if (lane == 0 && hasPos && hasNeg && hasSemi) {
            const float l = fmaxf(hp - shn + MARGIN, 0.f);
            atomicAdd(gsum, l);
            atomicAdd(gcnt, 1u);
        }
    }
}

// ---------------------------------------------------------------------------
// Kernel 3: finalize the scalar loss.
// ---------------------------------------------------------------------------
__global__ void tl_finalize(const float* gsum, const unsigned* gcnt, float* out) {
    const unsigned c = *gcnt;
    out[0] = c ? (*gsum) / (float)c : 0.f;
}

extern "C" void kernel_launch(void* const* d_in, const int* in_sizes, int n_in,
                              void* d_out, int out_size, void* d_ws, size_t ws_size,
                              hipStream_t stream) {
    const float* emb    = (const float*)d_in[0];   // [2048,1024] fp32
    const int*   labels = (const int*)d_in[1];     // [2048] int
    float*       out    = (float*)d_out;           // scalar fp32

    // Workspace layout: [0] float sum, [4] uint count, [64..] f16 normalized E (4MB).
    float*    gsum = (float*)d_ws;
    unsigned* gcnt = (unsigned*)((char*)d_ws + 4);
    half_t*   E    = (half_t*)((char*)d_ws + 64);

    tl_normalize<<<N_ROWS, 256, 0, stream>>>(emb, E, gsum, gcnt);
    tl_strip<<<N_ROWS / 16, 256, 0, stream>>>(E, labels, gsum, gcnt);
    tl_finalize<<<1, 1, 0, stream>>>(gsum, gcnt, out);
}
